// AttentionLayer_Topic_60730837565900
// MI455X (gfx1250) — matrix operand order repaired and checked by reading, not compiled, not run on hardware
//
#include <hip/hip_runtime.h>
#include <math.h>

// ---------------------------------------------------------------------------
// Types for CDNA5 WMMA (wave32): v_wmma_f32_16x16x32_bf16
// ---------------------------------------------------------------------------
typedef __attribute__((ext_vector_type(16))) __bf16 v16bf;
typedef __attribute__((ext_vector_type(8)))  __bf16 v8bf;
typedef __attribute__((ext_vector_type(4)))  __bf16 v4bf;
typedef __attribute__((ext_vector_type(8)))  float  v8f;
typedef __attribute__((ext_vector_type(4)))  int    v4i;

#define SQRT_HALF 0.70710678118654752440f

// CDNA5 async global->LDS copy (tracked by ASYNCcnt)
#if defined(__gfx1250__) && __has_builtin(__builtin_amdgcn_global_load_async_to_lds_b128)
#define HAVE_ASYNC_LDS 1
#else
#define HAVE_ASYNC_LDS 0
#endif

#if HAVE_ASYNC_LDS
typedef __attribute__((address_space(1))) v4i gv4i;   // global (__device__) int4
typedef __attribute__((address_space(3))) v4i lv4i;   // LDS (__shared__) int4
__device__ __forceinline__ void async_cp16(const __bf16* g, __bf16* l) {
    __builtin_amdgcn_global_load_async_to_lds_b128((gv4i*)g, (lv4i*)l, 0, 0);
}
__device__ __forceinline__ void wait_async_lds() {
#if __has_builtin(__builtin_amdgcn_s_wait_asynccnt)
    __builtin_amdgcn_s_wait_asynccnt(0);
#else
    asm volatile("s_wait_asynccnt 0" ::: "memory");
#endif
}
#endif

__device__ __forceinline__ v16bf cat8(v8bf lo, v8bf hi) {
    return __builtin_shufflevector(lo, hi, 0,1,2,3,4,5,6,7,8,9,10,11,12,13,14,15);
}

// ---------------------------------------------------------------------------
// Weight-norm prep: W[o][i] = bf16( v[o*I+i] * g[o] / ||v[o,:]|| ), row-major.
// Row-major W IS the n-major layout of the GEMM's B operand (W^T), so no
// transpose is needed anywhere for the weight matrices.
// ---------------------------------------------------------------------------
__global__ __launch_bounds__(256) void wn_prep_kernel(
    const float* __restrict__ v, const float* __restrict__ g,
    __bf16* __restrict__ W, int O, int I)
{
    __shared__ float red[256];
    const int o = blockIdx.x;
    const int t = threadIdx.x;
    float s = 0.f;
    for (int i = t; i < I; i += 256) {
        float x = v[(long)o * I + i];
        s += x * x;
    }
    red[t] = s;
    __syncthreads();
    for (int w = 128; w > 0; w >>= 1) {
        if (t < w) red[t] += red[t + w];
        __syncthreads();
    }
    const float sc = g[o] * rsqrtf(red[0]);
    for (int i = t; i < I; i += 256)
        W[(long)o * I + i] = (__bf16)(v[(long)o * I + i] * sc);
}

// ---------------------------------------------------------------------------
// fp32 -> bf16 linear converter (A operands)
// ---------------------------------------------------------------------------
__global__ __launch_bounds__(256) void cvt_kernel(
    const float* __restrict__ src, __bf16* __restrict__ dst, long n)
{
    long i = (long)blockIdx.x * blockDim.x + threadIdx.x;
    const long stride = (long)gridDim.x * blockDim.x;
    for (; i < n; i += stride) dst[i] = (__bf16)src[i];
}

// ---------------------------------------------------------------------------
// Tiled transpose + convert: dst[b][j][i] = bf16(a[b][i][j] (+ b2[b][i][j]))
// src is [R x Cc] per batch, dst is [Cc x R]. 32x32 LDS tile, coalesced both
// sides. Produces the n-major B operands for the batched GEMMs.
// ---------------------------------------------------------------------------
__global__ __launch_bounds__(256) void transpose_cvt_kernel(
    const float* __restrict__ a, const float* __restrict__ b2,
    __bf16* __restrict__ dst, int R, int Cc)
{
    __shared__ float tile[32][33];
    const long sOff = (long)blockIdx.z * R * Cc;
    const int j0 = blockIdx.x * 32;       // src col
    const int i0 = blockIdx.y * 32;       // src row
    const int tx = threadIdx.x & 31;
    const int ty = threadIdx.x >> 5;      // 0..7
#pragma unroll
    for (int r = 0; r < 4; ++r) {
        const int ii = ty + r * 8;
        float v = a[sOff + (long)(i0 + ii) * Cc + j0 + tx];
        if (b2) v += b2[sOff + (long)(i0 + ii) * Cc + j0 + tx];
        tile[ii][tx] = v;
    }
    __syncthreads();
#pragma unroll
    for (int r = 0; r < 4; ++r) {
        const int jj = ty + r * 8;
        dst[sOff + (long)(j0 + jj) * R + i0 + tx] = (__bf16)tile[tx][jj];
    }
}

// ---------------------------------------------------------------------------
// Tiled bf16 GEMM with fused epilogues. BM=BN=128, BK=32, 256 threads =
// 8 wave32 waves (4Mx2N), each wave a 32x64 tile = 2x4 WMMA accumulators.
// B operand is pre-stored n-major: Bp[n*ldb + k] -> both A and B tile loads
// are contiguous 16B copies, staged via async-to-LDS, double buffered.
// Epilogues:
//   EPI 0: h      = (acc + bias[col] + add[row,col]) * scale -> bf16
//   EPI 1: scores =  acc                                     -> f32
//   EPI 2: ctx    =  acc * scale                             -> bf16
//   EPI 3: out    = (acc + bias[col] + add[row,col]) * scale -> f32
// ---------------------------------------------------------------------------
#define BM 128
#define BN 128
#define BK 32
#define LDAS 40   // (BK + 8) bf16 -> 80B row stride, 16B-aligned
#define LDBS 40

template <int EPI>
__global__ __launch_bounds__(256) void gemm_bf16_kernel(
    const __bf16* __restrict__ A, const __bf16* __restrict__ Bp,
    int M, int N, int K, int lda, int ldb,
    long strideA, long strideB, long strideC,
    void* __restrict__ Cout, int ldc,
    const float* __restrict__ bias,
    const float* __restrict__ addf32, int ldadd,
    float scale)
{
    __shared__ __align__(16) __bf16 As[2][BM * LDAS];
    __shared__ __align__(16) __bf16 Bs[2][BN * LDBS];

    const int tid  = threadIdx.x;
    const int lane = tid & 31;
    const int wave = tid >> 5;        // 0..7
    const int wm   = wave >> 1;       // 0..3 -> M
    const int wn   = wave & 1;        // 0..1 -> N
    const int lane16   = lane & 15;
    const int laneHalf = lane >> 4;   // 0 | 1

    const int batch = blockIdx.z;
    A  += (long)batch * strideA;
    Bp += (long)batch * strideB;
    const long cbatch = (long)batch * strideC;

    const int tileM0 = blockIdx.y * BM;
    const int tileN0 = blockIdx.x * BN;

    v8f acc[2][4];
#pragma unroll
    for (int i = 0; i < 2; ++i)
#pragma unroll
        for (int j = 0; j < 4; ++j)
            acc[i][j] = (v8f){0.f,0.f,0.f,0.f,0.f,0.f,0.f,0.f};

    // tile loader mapping: 64 rows x 4 k-chunks per pass, two passes
    const int lRow = tid >> 2;        // 0..63
    const int lCol = (tid & 3) * 8;   // 0,8,16,24
    const __bf16* aSrc = A  + (long)(tileM0 + lRow) * lda + lCol;
    const __bf16* bSrc = Bp + (long)(tileN0 + lRow) * ldb + lCol;
    const long aStep64 = (long)64 * lda;
    const long bStep64 = (long)64 * ldb;
    const int  ldsOff  = lRow * LDAS + lCol;

#if !HAVE_ASYNC_LDS
    v8bf ra0, ra1, rb0, rb1;
#endif

    // ---- prologue: stage k0 = 0 into buffer 0
#if HAVE_ASYNC_LDS
    async_cp16(aSrc,           &As[0][ldsOff]);
    async_cp16(aSrc + aStep64, &As[0][ldsOff + 64 * LDAS]);
    async_cp16(bSrc,           &Bs[0][ldsOff]);
    async_cp16(bSrc + bStep64, &Bs[0][ldsOff + 64 * LDBS]);
    wait_async_lds();
#else
    ra0 = *(const v8bf*)(aSrc);
    ra1 = *(const v8bf*)(aSrc + aStep64);
    rb0 = *(const v8bf*)(bSrc);
    rb1 = *(const v8bf*)(bSrc + bStep64);
    *(v8bf*)&As[0][ldsOff]             = ra0;
    *(v8bf*)&As[0][ldsOff + 64 * LDAS] = ra1;
    *(v8bf*)&Bs[0][ldsOff]             = rb0;
    *(v8bf*)&Bs[0][ldsOff + 64 * LDBS] = rb1;
#endif
    __syncthreads();

    int cur = 0;
    for (int k0 = 0; k0 < K; k0 += BK, cur ^= 1) {
        const int nxt = cur ^ 1;
        const bool more = (k0 + BK) < K;

        // ---- stage next tile (overlaps with WMMA below)
        if (more) {
#if HAVE_ASYNC_LDS
            async_cp16(aSrc + k0 + BK,           &As[nxt][ldsOff]);
            async_cp16(aSrc + k0 + BK + aStep64, &As[nxt][ldsOff + 64 * LDAS]);
            async_cp16(bSrc + k0 + BK,           &Bs[nxt][ldsOff]);
            async_cp16(bSrc + k0 + BK + bStep64, &Bs[nxt][ldsOff + 64 * LDBS]);
#else
            ra0 = *(const v8bf*)(aSrc + k0 + BK);
            ra1 = *(const v8bf*)(aSrc + k0 + BK + aStep64);
            rb0 = *(const v8bf*)(bSrc + k0 + BK);
            rb1 = *(const v8bf*)(bSrc + k0 + BK + bStep64);
#endif
        }

        // ---- fragments per CDNA5 ISA layouts
        // A 16x32 bf16: lanes 0-15 K[0..7]&K[16..23] of row M=lane;
        //               lanes 16-31 K[8..15]&K[24..31].
        const __bf16* Ac = &As[cur][0];
        const __bf16* Bc = &Bs[cur][0];
        v16bf afrag[2];
#pragma unroll
        for (int i = 0; i < 2; ++i) {
            const int row = wm * 32 + i * 16 + lane16;
            const int kb  = laneHalf * 8;
            const v8bf lo = *(const v8bf*)(Ac + row * LDAS + kb);
            const v8bf hi = *(const v8bf*)(Ac + row * LDAS + kb + 16);
            afrag[i] = cat8(lo, hi);
        }
        // B 32x16 bf16: lanes 0-15 K[0..7]&K[16..23] of col N=lane; lanes 16-31 K[8..15]&K[24..31]
        v16bf bfrag[4];
#pragma unroll
        for (int j = 0; j < 4; ++j) {
            const int col = wn * 64 + j * 16 + lane16;
            const int kb  = laneHalf * 8;
            const v8bf lo = *(const v8bf*)(Bc + col * LDBS + kb);
            const v8bf hi = *(const v8bf*)(Bc + col * LDBS + kb + 16);
            bfrag[j] = cat8(lo, hi);
        }

#pragma unroll
        for (int i = 0; i < 2; ++i)
#pragma unroll
            for (int j = 0; j < 4; ++j)
                acc[i][j] = __builtin_amdgcn_wmma_f32_16x16x32_bf16(
                    false, afrag[i], false, bfrag[j],
                    (short)0, acc[i][j], false, false);

        if (more) {
#if HAVE_ASYNC_LDS
            wait_async_lds();
#else
            *(v8bf*)&As[nxt][ldsOff]             = ra0;
            *(v8bf*)&As[nxt][ldsOff + 64 * LDAS] = ra1;
            *(v8bf*)&Bs[nxt][ldsOff]             = rb0;
            *(v8bf*)&Bs[nxt][ldsOff + 64 * LDBS] = rb1;
#endif
        }
        __syncthreads();
    }

    // ---- epilogue: C/D layout -> element (r + laneHalf*8, lane16)
#pragma unroll
    for (int i = 0; i < 2; ++i) {
#pragma unroll
        for (int j = 0; j < 4; ++j) {
            const int col     = tileN0 + wn * 64 + j * 16 + lane16;
            const int rowBase = tileM0 + wm * 32 + i * 16 + laneHalf * 8;
#pragma unroll
            for (int r = 0; r < 8; ++r) {
                const int row = rowBase + r;
                float vv = acc[i][j][r];
                if constexpr (EPI == 0) {
                    vv = (vv + bias[col] + addf32[(long)row * ldadd + col]) * scale;
                    ((__bf16*)Cout)[cbatch + (long)row * ldc + col] = (__bf16)vv;
                } else if constexpr (EPI == 1) {
                    ((float*)Cout)[cbatch + (long)row * ldc + col] = vv;
                } else if constexpr (EPI == 2) {
                    ((__bf16*)Cout)[cbatch + (long)row * ldc + col] = (__bf16)(vv * scale);
                } else {
                    vv = (vv + bias[col] + addf32[(long)row * ldadd + col]) * scale;
                    ((float*)Cout)[cbatch + (long)row * ldc + col] = vv;
                }
            }
        }
    }
}

// ---------------------------------------------------------------------------
// Row softmax over S=1024 (256 threads x 4 contiguous elems), in-place f32
// plus a bf16 copy for the ctx GEMM.
// ---------------------------------------------------------------------------
__global__ __launch_bounds__(256) void softmax_kernel(
    float* __restrict__ sc, __bf16* __restrict__ ab, int S)
{
    __shared__ float red[256];
    const long row = blockIdx.x;
    float* p = sc + row * (long)S;
    const int t = threadIdx.x;

    float4 v = *(float4*)(p + t * 4);
    float m = fmaxf(fmaxf(v.x, v.y), fmaxf(v.z, v.w));
    red[t] = m;
    __syncthreads();
    for (int w = 128; w > 0; w >>= 1) {
        if (t < w) red[t] = fmaxf(red[t], red[t + w]);
        __syncthreads();
    }
    m = red[0];
    __syncthreads();

    float4 e;
    e.x = __expf(v.x - m); e.y = __expf(v.y - m);
    e.z = __expf(v.z - m); e.w = __expf(v.w - m);
    red[t] = e.x + e.y + e.z + e.w;
    __syncthreads();
    for (int w = 128; w > 0; w >>= 1) {
        if (t < w) red[t] += red[t + w];
        __syncthreads();
    }
    const float inv = 1.0f / red[0];
    e.x *= inv; e.y *= inv; e.z *= inv; e.w *= inv;

    *(float4*)(p + t * 4) = e;
    v4bf b; b[0] = (__bf16)e.x; b[1] = (__bf16)e.y; b[2] = (__bf16)e.z; b[3] = (__bf16)e.w;
    *(v4bf*)(ab + row * (long)S + t * 4) = b;
}

// ---------------------------------------------------------------------------
// Launch
// ---------------------------------------------------------------------------
extern "C" void kernel_launch(void* const* d_in, const int* in_sizes, int n_in,
                              void* d_out, int out_size, void* d_ws, size_t ws_size,
                              hipStream_t stream) {
    const float* x     = (const float*)d_in[0];
    const float* te    = (const float*)d_in[1];
    const float* enc0  = (const float*)d_in[2];
    const float* enc1  = (const float*)d_in[3];
    const float* enc2  = (const float*)d_in[4];
    const float* in_v  = (const float*)d_in[5];
    const float* in_g  = (const float*)d_in[6];
    const float* in_b  = (const float*)d_in[7];
    const float* out_v = (const float*)d_in[8];
    const float* out_g = (const float*)d_in[9];
    const float* out_b = (const float*)d_in[10];

    const int  B = 64, T = 1024, S = 1024, C = 512, E = 512;
    const long BT = (long)B * T;

    // workspace carve-up (bf16 staging, all 16B-aligned)
    char* ws = (char*)d_ws;
    __bf16* WIn  = (__bf16*)ws;  ws += (size_t)E * C * 2;      // [E x C] row-major (= n-major W^T)
    __bf16* WOut = (__bf16*)ws;  ws += (size_t)C * E * 2;      // [C x E] row-major
    __bf16* xb   = (__bf16*)ws;  ws += (size_t)BT * C * 2;     // [BT x C]
    __bf16* Kbt  = (__bf16*)ws;  ws += (size_t)B * S * E * 2;  // [B][S x E] = (enc0+enc1)^T
    __bf16* e2t  = (__bf16*)ws;  ws += (size_t)B * E * S * 2;  // [B][E x S] = enc2^T
    __bf16* hb   = (__bf16*)ws;  ws += (size_t)BT * E * 2;     // [BT x E]
    __bf16* ab   = (__bf16*)ws;  ws += (size_t)BT * S * 2;     // [BT x S]
    __bf16* ctxb = (__bf16*)ws;  ws += (size_t)BT * E * 2;     // [BT x E]

    float* outp  = (float*)d_out;           // [BT x C]
    float* attnF = outp + BT * (long)C;     // [BT x S]  (scores -> softmax in place)

    // 1) weight-norm -> bf16 row-major (already n-major for the GEMMs)
    wn_prep_kernel<<<E, 256, 0, stream>>>(in_v,  in_g,  WIn,  E, C);
    wn_prep_kernel<<<C, 256, 0, stream>>>(out_v, out_g, WOut, C, E);

    // 2) A-side convert + B-side transpose-converts
    cvt_kernel<<<4096, 256, 0, stream>>>(x, xb, BT * (long)C);
    transpose_cvt_kernel<<<dim3(S / 32, E / 32, B), 256, 0, stream>>>(
        enc0, enc1, Kbt, E, S);                       // [E x S] -> [S x E]
    transpose_cvt_kernel<<<dim3(E / 32, S / 32, B), 256, 0, stream>>>(
        enc2, nullptr, e2t, S, E);                    // [S x E] -> [E x S]

    // 3) h = (x @ Win^T + in_b + te) * sqrt(0.5)      [BT x E], bf16
    gemm_bf16_kernel<0><<<dim3(E / BN, BT / BM, 1), 256, 0, stream>>>(
        xb, WIn, (int)BT, E, C, C, C, 0, 0, 0,
        hb, E, in_b, te, E, SQRT_HALF);

    // 4) scores = h @ (enc0+enc1)   batched [T x S] -> f32 into attn region
    gemm_bf16_kernel<1><<<dim3(S / BN, T / BM, B), 256, 0, stream>>>(
        hb, Kbt, T, S, E, E, E,
        (long)T * E, (long)S * E, (long)T * S,
        attnF, S, nullptr, nullptr, 0, 1.0f);

    // 5) softmax rows (f32 in place + bf16 copy)
    softmax_kernel<<<(unsigned)BT, 256, 0, stream>>>(attnF, ab, S);

    // 6) ctx = attn @ enc2 * sqrt(S)   batched [T x E], bf16
    gemm_bf16_kernel<2><<<dim3(E / BN, T / BM, B), 256, 0, stream>>>(
        ab, e2t, T, E, S, S, S,
        (long)T * S, (long)E * S, (long)T * E,
        ctxb, E, nullptr, nullptr, 0, 32.0f /* sqrt(1024) */);

    // 7) out = (ctx @ Wout^T + out_b + x) * sqrt(0.5)   [BT x C], f32
    gemm_bf16_kernel<3><<<dim3(C / BN, BT / BM, 1), 256, 0, stream>>>(
        ctxb, WOut, (int)BT, C, E, E, E, 0, 0, 0,
        outp, C, out_b, x, C, SQRT_HALF);
}
